// MutiGAT_24910810316937
// MI455X (gfx1250) — matrix operand order repaired and checked by reading, not compiled, not run on hardware
//
#include <hip/hip_runtime.h>
#include <math.h>

typedef __attribute__((ext_vector_type(16))) _Float16 v16h;
typedef __attribute__((ext_vector_type(8)))  float    v8f;

union FragV { unsigned int u[8]; v16h v; };

// ---------------- helpers ----------------
__device__ __forceinline__ unsigned fenc(float f) {
    unsigned u = __float_as_uint(f);
    return (u & 0x80000000u) ? ~u : (u | 0x80000000u);
}
__device__ __forceinline__ float fdec(unsigned k) {
    unsigned u = (k & 0x80000000u) ? (k ^ 0x80000000u) : ~k;
    return __uint_as_float(u);
}
__device__ __forceinline__ void edge_sd(int e, const int* edge, int E, int& s, int& d) {
    if (e < E) { s = edge[e]; d = edge[E + e]; } else { s = d = e - E; }
}
__device__ __forceinline__ float lrelu(float v, float sl) { return v > 0.f ? v : sl * v; }

// ---------------- utility kernels ----------------
__global__ void k_zero_f(float* a, int n) {
    int i = blockIdx.x * blockDim.x + threadIdx.x;
    if (i < n) a[i] = 0.f;
}
__global__ void k_fill_u32(unsigned* a, unsigned v, int n) {
    int i = blockIdx.x * blockDim.x + threadIdx.x;
    if (i < n) a[i] = v;
}
__global__ void k_f2h(const float* __restrict__ in, _Float16* __restrict__ out, int n) {
    int i = blockIdx.x * blockDim.x + threadIdx.x;
    if (i < n) out[i] = (_Float16)in[i];
}
__global__ void k_mask_count(const unsigned char* mask, float* msum, int n) {
    int i = blockIdx.x * blockDim.x + threadIdx.x;
    if (i < n && mask[i]) atomicAdd(msum, 1.0f);
}

// ---------------- WMMA GEMM: C[nrows,ncols] = A[nrows,K](f16) * B[ncols,K](f16)^T ----------------
#define BM 128
#define BN 64
#define BK 32
__global__ __launch_bounds__(256) void k_gemm_wmma(
    const _Float16* __restrict__ A, const _Float16* __restrict__ B,
    float* __restrict__ C, int nrows, int K, int ncols)
{
    __shared__ __align__(16) _Float16 As[BM * BK];
    __shared__ __align__(16) _Float16 Bs[BN * BK];
    const int tid  = threadIdx.x;
    const int lane = tid & 31;
    const int wid  = tid >> 5;
    const int wm   = wid & 3;   // 4 waves along M
    const int wn   = wid >> 2;  // 2 waves along N
    const int bm = blockIdx.y, bn = blockIdx.x;

    v8f acc[2][2] = {};

    int arow = bm * BM + (tid >> 1);
    if (arow >= nrows) arow = nrows - 1;       // clamp loads; stores are guarded
    const int ahalf = (tid & 1) * 16;
    const int brow  = bn * BN + ((tid & 127) >> 1);
    const int bhalf = (tid & 1) * 16;

    for (int k0 = 0; k0 < K; k0 += BK) {
        // stage A tile (128x32) and B tile (64x32) into LDS
        {
            const uint4* sa = (const uint4*)&A[(size_t)arow * K + k0 + ahalf];
            uint4* da = (uint4*)&As[(tid >> 1) * BK + ahalf];
            da[0] = sa[0]; da[1] = sa[1];
            if (tid < 128) {
                const uint4* sb = (const uint4*)&B[(size_t)brow * K + k0 + bhalf];
                uint4* db = (uint4*)&Bs[((tid & 127) >> 1) * BK + bhalf];
                db[0] = sb[0]; db[1] = sb[1];
            }
            if (k0 + BK < K) {
                __builtin_prefetch(&A[(size_t)arow * K + k0 + BK + ahalf], 0, 1);
                if (tid < 128)
                    __builtin_prefetch(&B[(size_t)brow * K + k0 + BK + bhalf], 0, 1);
            }
        }
        __syncthreads();

        // build fragments per CDNA5 WMMA VGPR layouts
        FragV fa[2], fb[2];
        const int r  = lane & 15;
        const int ka = (lane < 16) ? 0 : 8;    // A: lanes<16 hold K{0..7,16..23}
        #pragma unroll
        for (int sm = 0; sm < 2; ++sm) {
            const _Float16* base = &As[(wm * 32 + sm * 16 + r) * BK];
            *(uint4*)&fa[sm].u[0] = *(const uint4*)&base[ka];
            *(uint4*)&fa[sm].u[4] = *(const uint4*)&base[ka + 16];
        }
        const int kb = (lane < 16) ? 0 : 8;    // B: lanes<16 hold K{0..7,16..23}
        #pragma unroll
        for (int sn = 0; sn < 2; ++sn) {
            const _Float16* base = &Bs[(wn * 32 + sn * 16 + r) * BK];
            *(uint4*)&fb[sn].u[0] = *(const uint4*)&base[kb];
            *(uint4*)&fb[sn].u[4] = *(const uint4*)&base[kb + 16];
        }
        #pragma unroll
        for (int sm = 0; sm < 2; ++sm)
            #pragma unroll
            for (int sn = 0; sn < 2; ++sn)
                acc[sm][sn] = __builtin_amdgcn_wmma_f32_16x16x32_f16(
                    false, fa[sm].v, false, fb[sn].v,
                    (short)0, acc[sm][sn], false, false);
        __syncthreads();
    }

    // store C: lanes 0-15 -> M=r(0..7), lanes 16-31 -> M=8+r; N = lane&15
    #pragma unroll
    for (int sm = 0; sm < 2; ++sm) {
        int row0 = bm * BM + wm * 32 + sm * 16 + ((lane < 16) ? 0 : 8);
        #pragma unroll
        for (int sn = 0; sn < 2; ++sn) {
            int col = bn * BN + wn * 32 + sn * 16 + (lane & 15);
            #pragma unroll
            for (int rr = 0; rr < 8; ++rr) {
                int rw = row0 + rr;
                if (rw < nrows) C[(size_t)rw * ncols + col] = acc[sm][sn][rr];
            }
        }
    }
}

// ---------------- attention kernels ----------------
__global__ void k_att_scores(const float* __restrict__ xh,
                             const float* __restrict__ as, const float* __restrict__ ad,
                             float* __restrict__ es, float* __restrict__ ed,
                             int n, int H, int C) {
    int idx = blockIdx.x * blockDim.x + threadIdx.x;
    if (idx >= n * H) return;
    int node = idx / H, h = idx % H;
    const float* row = xh + (size_t)node * H * C + (size_t)h * C;
    const float* a1 = as + h * C;
    const float* a2 = ad + h * C;
    float s1 = 0.f, s2 = 0.f;
    for (int c = 0; c < C; ++c) { float v = row[c]; s1 += v * a1[c]; s2 += v * a2[c]; }
    es[idx] = s1; ed[idx] = s2;
}

__global__ void k_edge_max(const float* __restrict__ es, const float* __restrict__ ed,
                           const int* __restrict__ edge, int E, int total, int H,
                           unsigned* __restrict__ menc) {
    int idx = blockIdx.x * blockDim.x + threadIdx.x;
    if (idx >= total * H) return;
    int e = idx / H, h = idx % H, s, d;
    edge_sd(e, edge, E, s, d);
    float v = lrelu(es[s * H + h] + ed[d * H + h], 0.2f);
    atomicMax(&menc[d * H + h], fenc(v));
}

__global__ void k_edge_sum(const float* __restrict__ es, const float* __restrict__ ed,
                           const int* __restrict__ edge, int E, int total, int H,
                           const unsigned* __restrict__ menc, float* __restrict__ den) {
    int idx = blockIdx.x * blockDim.x + threadIdx.x;
    if (idx >= total * H) return;
    int e = idx / H, h = idx % H, s, d;
    edge_sd(e, edge, E, s, d);
    float v = lrelu(es[s * H + h] + ed[d * H + h], 0.2f);
    atomicAdd(&den[d * H + h], expf(v - fdec(menc[d * H + h])));
}

__global__ void k_edge_alpha(const float* __restrict__ es, const float* __restrict__ ed,
                             const int* __restrict__ edge, int E, int total, int H,
                             const unsigned* __restrict__ menc, const float* __restrict__ den,
                             float* __restrict__ alpha) {
    int idx = blockIdx.x * blockDim.x + threadIdx.x;
    if (idx >= total * H) return;
    int e = idx / H, h = idx % H, s, d;
    edge_sd(e, edge, E, s, d);
    float v = lrelu(es[s * H + h] + ed[d * H + h], 0.2f);
    alpha[idx] = expf(v - fdec(menc[d * H + h])) / den[d * H + h];
}

__global__ void k_edge_agg(const float* __restrict__ feat, const float* __restrict__ alpha,
                           const int* __restrict__ edge, int E, int tot_elems, int H, int C,
                           float* __restrict__ out) {
    int idx = blockIdx.x * blockDim.x + threadIdx.x;
    if (idx >= tot_elems) return;
    int HC = H * C;
    int e = idx / HC, r = idx % HC, s, d;
    edge_sd(e, edge, E, s, d);
    atomicAdd(&out[(size_t)d * HC + r], alpha[e * H + r / C] * feat[(size_t)s * HC + r]);
}

// ---------------- layer glue kernels ----------------
__global__ void k_head_mean_bias(const float* __restrict__ acc3, const float* __restrict__ b,
                                 float* __restrict__ out, int n, int H, int C) {
    int idx = blockIdx.x * blockDim.x + threadIdx.x;
    if (idx >= n * C) return;
    int node = idx / C, c = idx % C;
    float s = 0.f;
    for (int h = 0; h < 3; ++h) s += acc3[(size_t)node * H * C + h * C + c];
    out[idx] = s * (1.f / 3.f) + b[c];
}

__global__ __launch_bounds__(256) void k_bn_stats(const float* __restrict__ hid, int n, int C,
                                                  float* __restrict__ mu, float* __restrict__ rstd) {
    __shared__ float ss[256], ss2[256];
    int c = blockIdx.x, t = threadIdx.x;
    float s = 0.f, s2 = 0.f;
    for (int i = t; i < n; i += 256) {
        float v = hid[(size_t)i * C + c];
        s += v; s2 += v * v;
    }
    ss[t] = s; ss2[t] = s2; __syncthreads();
    for (int o = 128; o > 0; o >>= 1) {
        if (t < o) { ss[t] += ss[t + o]; ss2[t] += ss2[t + o]; }
        __syncthreads();
    }
    if (t == 0) {
        float m = ss[0] / n;
        float var = ss2[0] / n - m * m;
        mu[c] = m;
        rstd[c] = rsqrtf(var + 1e-5f);
    }
}

__global__ void k_layer2(const float* __restrict__ hid, const float* __restrict__ mu,
                         const float* __restrict__ rstd, const float* __restrict__ gma,
                         const float* __restrict__ bta, const float* __restrict__ W2,
                         float* __restrict__ out2h, int n) {
    int node = blockIdx.x * blockDim.x + threadIdx.x;
    if (node >= n) return;
    float acc[6] = {0.f, 0.f, 0.f, 0.f, 0.f, 0.f};
    const float* row = hid + (size_t)node * 128;
    for (int c = 0; c < 128; ++c) {
        float v = lrelu((row[c] - mu[c]) * rstd[c] * gma[c] + bta[c], 0.01f);
        #pragma unroll
        for (int j = 0; j < 6; ++j) acc[j] += v * W2[j * 128 + c];
    }
    #pragma unroll
    for (int j = 0; j < 6; ++j) out2h[(size_t)node * 6 + j] = acc[j];
}

__global__ void k_finalize(const float* __restrict__ o2acc, const float* __restrict__ b2,
                           const int* __restrict__ y, const unsigned char* __restrict__ mask,
                           float* __restrict__ acc_out, float* __restrict__ scal, int n) {
    int node = blockIdx.x * blockDim.x + threadIdx.x;
    if (node >= n) return;
    const float* a = o2acc + (size_t)node * 6;
    float o0 = (a[0] + a[2] + a[4]) * (1.f / 3.f) + b2[0];
    float o1 = (a[1] + a[3] + a[5]) * (1.f / 3.f) + b2[1];
    float m = fmaxf(o0, o1);
    float lse = m + logf(expf(o0 - m) + expf(o1 - m));
    float lp0 = o0 - lse, lp1 = o1 - lse;
    acc_out[node] += expf(lp1) * (1.f / 3.f);
    if (mask[node]) atomicAdd(&scal[0], -(y[node] == 1 ? lp1 : lp0));
}

__global__ void k_write_loss(const float* scal, float* out, int n) {
    if (blockIdx.x == 0 && threadIdx.x == 0)
        out[n] = scal[0] / (scal[1] * 3.0f);
}

// ---------------- launch ----------------
extern "C" void kernel_launch(void* const* d_in, const int* in_sizes, int n_in,
                              void* d_out, int out_size, void* d_ws, size_t ws_size,
                              hipStream_t stream) {
    const int Nn = 20000, Fd = 2048, Ed = 320000, ET = Ed + Nn;
    const int H = 3, C1 = 128, NH1 = 384, C2 = 2, NH2 = 6, K = 2048;

    const float* x = (const float*)d_in[0];
    const int* yv = (const int*)d_in[1];
    const unsigned char* mask = (const unsigned char*)d_in[2];
    const int* eppi = (const int*)d_in[3];
    const int* ehom = (const int*)d_in[4];
    // d_in[5] (x_cc), d_in[6] (edge_cc), cc_params (17..26): exact no-op (softmax over
    // a singleton axis == 1.0), so the cellChat branch is skipped entirely.
    const float* W1  = (const float*)d_in[7];
    const float* as1 = (const float*)d_in[8];
    const float* ad1 = (const float*)d_in[9];
    const float* b1  = (const float*)d_in[10];
    const float* gma = (const float*)d_in[11];
    const float* bta = (const float*)d_in[12];
    const float* W2  = (const float*)d_in[13];
    const float* as2 = (const float*)d_in[14];
    const float* ad2 = (const float*)d_in[15];
    const float* b2  = (const float*)d_in[16];
    float* out = (float*)d_out;

    char* p = (char*)d_ws;
    auto carve = [&](size_t bytes) -> void* {
        void* r = (void*)p;
        p += (bytes + 255) & ~(size_t)255;
        return r;
    };
    _Float16* x16 = (_Float16*)carve((size_t)Nn * Fd * 2);
    _Float16* w16 = (_Float16*)carve((size_t)NH1 * Fd * 2);
    float* xh    = (float*)carve((size_t)Nn * NH1 * 4);
    float* oacc  = (float*)carve((size_t)Nn * NH1 * 4);
    float* hid   = (float*)carve((size_t)Nn * C1 * 4);
    float* es    = (float*)carve((size_t)Nn * H * 4);
    float* edp   = (float*)carve((size_t)Nn * H * 4);
    unsigned* menc = (unsigned*)carve((size_t)Nn * H * 4);
    float* den   = (float*)carve((size_t)Nn * H * 4);
    float* alpha = (float*)carve((size_t)ET * H * 4);
    float* o2h   = (float*)carve((size_t)Nn * NH2 * 4);
    float* o2acc = (float*)carve((size_t)Nn * NH2 * 4);
    float* mu    = (float*)carve(C1 * 4);
    float* rstd  = (float*)carve(C1 * 4);
    float* scal  = (float*)carve(2 * 4);  // [0]=loss numerator, [1]=mask count

    const int TB = 256;
    auto g = [&](int t) { return dim3((unsigned)((t + TB - 1) / TB)); };
    const unsigned ENC_NEG_MAX = 0x00800000u;  // fenc(-FLT_MAX)

    k_f2h<<<g(Nn * Fd), TB, 0, stream>>>(x, x16, Nn * Fd);
    k_zero_f<<<g(Nn + 1), TB, 0, stream>>>(out, Nn + 1);
    k_zero_f<<<1, TB, 0, stream>>>(scal, 2);
    k_mask_count<<<g(Nn), TB, 0, stream>>>(mask, scal + 1, Nn);

    for (int i = 0; i < 3; ++i) {
        const int* eg = (i % 2 == 0) ? eppi : ehom;

        // Layer 1: WMMA GEMM  xh = x @ W1_i^T
        k_f2h<<<g(NH1 * Fd), TB, 0, stream>>>(W1 + (size_t)i * NH1 * Fd, w16, NH1 * Fd);
        k_gemm_wmma<<<dim3(NH1 / BN, (Nn + BM - 1) / BM), 256, 0, stream>>>(
            x16, w16, xh, Nn, K, NH1);

        // Attention softmax (layer 1)
        k_att_scores<<<g(Nn * H), TB, 0, stream>>>(xh, as1 + i * H * C1, ad1 + i * H * C1,
                                                   es, edp, Nn, H, C1);
        k_fill_u32<<<g(Nn * H), TB, 0, stream>>>(menc, ENC_NEG_MAX, Nn * H);
        k_zero_f<<<g(Nn * H), TB, 0, stream>>>(den, Nn * H);
        k_edge_max<<<g(ET * H), TB, 0, stream>>>(es, edp, eg, Ed, ET, H, menc);
        k_edge_sum<<<g(ET * H), TB, 0, stream>>>(es, edp, eg, Ed, ET, H, menc, den);
        k_edge_alpha<<<g(ET * H), TB, 0, stream>>>(es, edp, eg, Ed, ET, H, menc, den, alpha);
        k_zero_f<<<g(Nn * NH1), TB, 0, stream>>>(oacc, Nn * NH1);
        k_edge_agg<<<g(ET * NH1), TB, 0, stream>>>(xh, alpha, eg, Ed, ET * NH1, H, C1, oacc);
        k_head_mean_bias<<<g(Nn * C1), TB, 0, stream>>>(oacc, b1 + i * C1, hid, Nn, H, C1);

        // BN + leaky + layer-2 linear
        k_bn_stats<<<dim3(C1), 256, 0, stream>>>(hid, Nn, C1, mu, rstd);
        k_layer2<<<g(Nn), TB, 0, stream>>>(hid, mu, rstd, gma + i * C1, bta + i * C1,
                                           W2 + (size_t)i * NH2 * C1, o2h, Nn);

        // Attention softmax (layer 2)
        k_att_scores<<<g(Nn * H), TB, 0, stream>>>(o2h, as2 + i * H * C2, ad2 + i * H * C2,
                                                   es, edp, Nn, H, C2);
        k_fill_u32<<<g(Nn * H), TB, 0, stream>>>(menc, ENC_NEG_MAX, Nn * H);
        k_zero_f<<<g(Nn * H), TB, 0, stream>>>(den, Nn * H);
        k_edge_max<<<g(ET * H), TB, 0, stream>>>(es, edp, eg, Ed, ET, H, menc);
        k_edge_sum<<<g(ET * H), TB, 0, stream>>>(es, edp, eg, Ed, ET, H, menc, den);
        k_edge_alpha<<<g(ET * H), TB, 0, stream>>>(es, edp, eg, Ed, ET, H, menc, den, alpha);
        k_zero_f<<<g(Nn * NH2), TB, 0, stream>>>(o2acc, Nn * NH2);
        k_edge_agg<<<g(ET * NH2), TB, 0, stream>>>(o2h, alpha, eg, Ed, ET * NH2, H, C2, o2acc);

        // log-softmax, acc, loss
        k_finalize<<<g(Nn), TB, 0, stream>>>(o2acc, b2 + i * C2, yv, mask, out, scal, Nn);
    }
    k_write_loss<<<1, 1, 0, stream>>>(scal, out, Nn);
}
